// WindowAttentionV2_56401510531723
// MI455X (gfx1250) — matrix operand order repaired and checked by reading, not compiled, not run on hardware
//
#include <hip/hip_runtime.h>
#include <hip/hip_bf16.h>
#include <cstdint>
#include <cstddef>

typedef __attribute__((ext_vector_type(16))) _Float16 v16h;
typedef __attribute__((ext_vector_type(8)))  float    v8f;
typedef unsigned int u32x4 __attribute__((ext_vector_type(4)));
typedef int          i32x8 __attribute__((ext_vector_type(8)));
typedef int          i32x4 __attribute__((ext_vector_type(4)));

#define DEVINL __device__ __forceinline__

#if defined(__has_builtin)
#  if __has_builtin(__builtin_amdgcn_tensor_load_to_lds) && \
      __has_builtin(__builtin_amdgcn_s_wait_tensorcnt)
#    define HAVE_TDM 1
#  endif
#endif
#ifndef HAVE_TDM
#  define HAVE_TDM 0
#endif

constexpr int BW   = 4096;   // windows
constexpr int NTOK = 64;     // tokens per window
constexpr int CH   = 384;    // channels
constexpr int NH   = 12;     // heads
constexpr int HD   = 32;     // head dim
constexpr int C3   = 1152;   // 3*C
constexpr int HID  = 64;     // mlp hidden

// workspace layout (bytes)
constexpr size_t BIAS_OFF   = 0;
constexpr size_t BIAS_BYTES = (size_t)NH * NTOK * NTOK * 4;        // 196608 (256-aligned)
constexpr size_t QKV_OFF    = BIAS_OFF + BIAS_BYTES;
constexpr size_t QKV_BYTES  = (size_t)BW * NTOK * C3 * 2;          // f16
constexpr size_t AO_OFF     = QKV_OFF + QKV_BYTES;

// ---------------- WMMA fragment helpers (CDNA5 ISA §7.12.2) ----------------

// 16-bit A matrix 16x32 (MxK), LDS row-major with `stride` halfs per row.
// lanes 0-15: M=lane, VGPR0-3 K=0..7, VGPR4-7 K=16..23; lanes 16-31: K=8..15 / 24..31.
DEVINL v16h frag_a(const _Float16* p, int stride, int lane) {
  const int m = lane & 15, hf = lane >> 4;
  const _Float16* row = p + m * stride;
  v16h a;
#pragma unroll
  for (int j = 0; j < 4; ++j) {
    a[2*j]       = row[hf*8 + 2*j];
    a[2*j + 1]   = row[hf*8 + 2*j + 1];
    a[8 + 2*j]   = row[16 + hf*8 + 2*j];
    a[8 + 2*j+1] = row[16 + hf*8 + 2*j + 1];
  }
  return a;
}

// 16-bit B matrix 32x16 (KxN) built from a TRANSPOSED LDS image bt[n][k] (stride halfs).
// lane = N (mod 16); lanes 0-15 hold K=0..7,16..23, lanes 16-31 hold K=8..15,24..31, 2 per VGPR.
DEVINL v16h frag_b(const _Float16* bt, int stride, int lane) {
  const int n = lane & 15, kg = lane >> 4;
  const _Float16* row = bt + n * stride + kg * 8;
  v16h b;
#pragma unroll
  for (int j = 0; j < 16; ++j) b[j] = row[j + (j >> 3) * 8];
  return b;
}

DEVINL v8f wmma_f16(v16h a, v16h b, v8f c) {
  return __builtin_amdgcn_wmma_f32_16x16x32_f16(false, a, false, b, (short)0, c,
                                                false, false);
}

DEVINL v8f vzero8() {
  v8f z;
#pragma unroll
  for (int i = 0; i < 8; ++i) z[i] = 0.f;
  return z;
}

#if HAVE_TDM
// LDS byte offset of a __shared__ object: flat shared-aperture addresses carry the
// LDS offset in the low 32 bits (CDNA5 ISA aperture truncation rule).
DEVINL uint32_t lds_addr_of(const void* p) { return (uint32_t)(uintptr_t)p; }

// TDM 2D fp32 tile load: tensor == tile (rows x cols), contiguous rows.
// D# layout per CDNA5 ISA.
// This toolchain exposes the 6-arg builtin:
//   (uint32x4 g0, int32x8 g1, int32x4 g2, int32x4 g3, int32x8 extra, i32 cpol)
DEVINL void tdm_load_2d_f32(uint32_t lds_off, const float* gptr, int rows, int cols) {
  uint64_t ga = (uint64_t)(uintptr_t)gptr;
  u32x4 g0;
  g0[0] = 1u;                                   // count=1, user mode, no gather
  g0[1] = lds_off;                              // lds_addr (bytes)
  g0[2] = (uint32_t)ga;                         // global_addr[31:0]
  g0[3] = (uint32_t)((ga >> 32) & 0x1FFFFFFu)   // global_addr[56:32]
          | (2u << 30);                         // type=2 ("image")
  i32x8 g1;
  g1[0] = 2 << 16;                              // data_size=4B; mask=0; no iterate/pad
  g1[1] = (cols & 0xFFFF) << 16;                // tensor_dim0[15:0]
  g1[2] = ((cols >> 16) & 0xFFFF) | ((rows & 0xFFFF) << 16);  // dim0 hi | dim1 lo
  g1[3] = ((rows >> 16) & 0xFFFF) | ((cols & 0xFFFF) << 16);  // dim1 hi | tile_dim0
  g1[4] = rows & 0xFFFF;                        // tile_dim1 (tile_dim2=0 -> unused)
  g1[5] = cols;                                 // tensor_dim0_stride[31:0]
  g1[6] = 0;                                    // stride hi | dim1_stride lo
  g1[7] = 0;                                    // dim1_stride hi
  i32x4 z4;
  z4[0] = 0; z4[1] = 0; z4[2] = 0; z4[3] = 0;   // groups 2/3: higher dims unused
  i32x8 z8;
#pragma unroll
  for (int i = 0; i < 8; ++i) z8[i] = 0;
  __builtin_amdgcn_tensor_load_to_lds(g0, g1, z4, z4, z8, 0);
}
#endif

// ---------------- Kernel 1: continuous relative position bias MLP ----------------
// bias_ws[h*4096 + i*64 + j]  (H, N, N) fp32 — computed ONCE (shared by all windows).
__global__ void crpb_bias_kernel(const float* __restrict__ w1, const float* __restrict__ b1,
                                 const float* __restrict__ w2, const float* __restrict__ b2,
                                 float* __restrict__ bias) {
  int p = blockIdx.x * blockDim.x + threadIdx.x;
  if (p >= NTOK * NTOK) return;
  int i = p >> 6, j = p & 63;
  float d0 = (float)((i >> 3) - (j >> 3));
  float d1 = (float)((i & 7) - (j & 7));
  float r0 = (d0 > 0.f ? 1.f : (d0 < 0.f ? -1.f : 0.f)) * log1pf(fabsf(d0));
  float r1 = (d1 > 0.f ? 1.f : (d1 < 0.f ? -1.f : 0.f)) * log1pf(fabsf(d1));
  float acc[NH];
#pragma unroll
  for (int hh = 0; hh < NH; ++hh) acc[hh] = b2[hh];
  for (int k = 0; k < HID; ++k) {
    float hv = fmaxf(r0 * w1[k] + r1 * w1[HID + k] + b1[k], 0.f);
#pragma unroll
    for (int hh = 0; hh < NH; ++hh) acc[hh] += hv * w2[k * NH + hh];
  }
#pragma unroll
  for (int hh = 0; hh < NH; ++hh) bias[hh * (NTOK * NTOK) + p] = acc[hh];
}

// ---------------- Kernel 2/4: 64x64-tile GEMM, f16 WMMA, f32 accumulate ----------------
// C[row0:row0+64, col0:col0+64] = A(rows x K) * W(K x ldw) + bias(row vec)
// grid = (colTiles, rowTiles): consecutive blocks share the A row-tile -> L2 reuse,
// A streams from HBM ~once instead of colTiles times.
template <bool AHALF, bool OUTHALF>
__global__ void gemm64_kernel(const void* __restrict__ Aptr, const float* __restrict__ W,
                              const float* __restrict__ bv, void* __restrict__ Out,
                              int K, int ldw) {
  __shared__ _Float16 As[64 * 36];   // A chunk 64x32, padded stride 36
  __shared__ _Float16 Bt[64 * 36];   // W chunk transposed: Bt[n][k], 64x32
  __shared__ float    bs[64];
  const int tid = threadIdx.x;
  const int lane = tid & 31, wv = tid >> 5;
  const int  col0 = blockIdx.x * 64;
  const long row0 = (long)blockIdx.y * 64;
  if (tid < 64) bs[tid] = bv[col0 + tid];
  v8f acc[4];
#pragma unroll
  for (int t = 0; t < 4; ++t) acc[t] = vzero8();

  const float*    Af = (const float*)Aptr;
  const _Float16* Ah = (const _Float16*)Aptr;

  for (int k0 = 0; k0 < K; k0 += 32) {
    __syncthreads();
#pragma unroll
    for (int i = 0; i < 16; ++i) {
      int e = i * 128 + tid;
      int r = e >> 5, c = e & 31;          // A chunk: 64 rows x 32 cols
      float xv = AHALF ? (float)Ah[(row0 + r) * (long)K + k0 + c]
                       : Af[(row0 + r) * (long)K + k0 + c];
      As[r * 36 + c] = (_Float16)xv;
      int kr = e >> 6, cc = e & 63;        // W chunk: 32 rows(K) x 64 cols -> transposed
      Bt[cc * 36 + kr] = (_Float16)W[(long)(k0 + kr) * ldw + col0 + cc];
    }
    __syncthreads();
    v16h a = frag_a(As + wv * 16 * 36, 36, lane);
#pragma unroll
    for (int t = 0; t < 4; ++t) {
      v16h b = frag_b(Bt + t * 16 * 36, 36, lane);
      acc[t] = wmma_f16(a, b, acc[t]);
    }
  }

  const int hf = lane >> 4, nl = lane & 15;
#pragma unroll
  for (int t = 0; t < 4; ++t) {
    int cg = col0 + t * 16 + nl;
    float badd = bs[t * 16 + nl];
#pragma unroll
    for (int r = 0; r < 8; ++r) {
      long rg = row0 + wv * 16 + r + 8 * hf;   // C layout: M = r + 8*(lane>=16)
      float v = acc[t][r] + badd;
      if (OUTHALF) ((_Float16*)Out)[rg * ldw + cg] = (_Float16)v;
      else         ((float*)Out)[rg * ldw + cg]    = v;
    }
  }
}

// ---------------- Kernel 3: fused cosine attention per (window, head) ----------------
// grid (NH, BW): the 12 heads sharing one window's mask tile are adjacent -> L2 reuse.
// 128 threads = 4 waves; wave wv owns query rows [16*wv, 16*wv+16).
__global__ void attn_kernel(const _Float16* __restrict__ qkv, const float* __restrict__ mask,
                            const float* __restrict__ bias, const float* __restrict__ tau,
                            _Float16* __restrict__ ao) {
  __shared__ _Float16 qs[64 * 36];     // q-hat, row-major [tok][dim]
  __shared__ _Float16 ks[64 * 36];     // k-hat, row-major [tok][dim] (== B^T for S = q k^T)
  __shared__ _Float16 vt[32 * 72];     // v transposed: vt[dim][tok]  (== B^T for O = P v)
  __shared__ _Float16 ps[4 * 16 * 68]; // per-wave P block, row-major [row][64]
  __shared__ float    ms_m[64 * 64];   // mask[win] tile (TDM-staged)
  __shared__ float    bs_m[64 * 64];   // bias[h] tile   (TDM-staged)
  const int h = blockIdx.x, win = blockIdx.y;
  const int tid = threadIdx.x, lane = tid & 31, wv = tid >> 5;

  const float* mw = mask + (size_t)win * (NTOK * NTOK);
  const float* bh = bias + (size_t)h * (NTOK * NTOK);

#if HAVE_TDM
  // One wave drives the Tensor Data Mover: two 16KB 2D-tile DMAs into LDS.
  if (wv == 0) {
    tdm_load_2d_f32(lds_addr_of(ms_m), mw, NTOK, NTOK);
    tdm_load_2d_f32(lds_addr_of(bs_m), bh, NTOK, NTOK);
    __builtin_amdgcn_s_wait_tensorcnt(0);
  }
#else
  // fallback: cooperative vectorized copy
#pragma unroll
  for (int i = 0; i < 8; ++i) {
    int e = (i * 128 + tid) * 4;
    *(float4*)&ms_m[e] = *(const float4*)&mw[e];
    *(float4*)&bs_m[e] = *(const float4*)&bh[e];
  }
#endif

  // ---- load q/k/v for this (window, head); L2-normalize q,k in fp32 ----
  {
    int row = tid >> 1, part = tid & 1;  // 2 threads per token row, 16 dims each
    const _Float16* base = qkv + ((size_t)win * NTOK + row) * C3 + h * HD + part * 16;
    float qv[16], kv[16], vv[16];
    float ssq = 0.f, ssk = 0.f;
#pragma unroll
    for (int j = 0; j < 16; ++j) { qv[j] = (float)base[j];        ssq += qv[j] * qv[j]; }
#pragma unroll
    for (int j = 0; j < 16; ++j) { kv[j] = (float)base[CH + j];   ssk += kv[j] * kv[j]; }
#pragma unroll
    for (int j = 0; j < 16; ++j) { vv[j] = (float)base[2 * CH + j]; }
    ssq += __shfl_xor(ssq, 1, 32);       // partner thread holds other 16 dims
    ssk += __shfl_xor(ssk, 1, 32);
    float iq = 1.f / fmaxf(sqrtf(ssq), 1e-12f);
    float ik = 1.f / fmaxf(sqrtf(ssk), 1e-12f);
#pragma unroll
    for (int j = 0; j < 16; ++j) {
      qs[row * 36 + part * 16 + j]   = (_Float16)(qv[j] * iq);
      ks[row * 36 + part * 16 + j]   = (_Float16)(kv[j] * ik);
      vt[(part * 16 + j) * 72 + row] = (_Float16)vv[j];
    }
  }
  __syncthreads();   // publishes q/k/v staging AND the TDM tiles to all waves

  const float tauv = tau[h];
  const int hf = lane >> 4, nl = lane & 15;

  // ---- S = q-hat @ k-hat^T : wave does 16x64 strip as 4 WMMA (K=32 in one op) ----
  v8f s[4];
  v16h aq = frag_a(qs + wv * 16 * 36, 36, lane);
#pragma unroll
  for (int t = 0; t < 4; ++t) {
    v16h bk = frag_b(ks + t * 16 * 36, 36, lane);
    s[t] = wmma_f16(aq, bk, vzero8());
  }

  // ---- tau scale + bias + mask (from LDS), row softmax (xor-shuffle reductions) ----
#pragma unroll
  for (int r = 0; r < 8; ++r) {
    int mg = wv * 16 + r + 8 * hf;       // query row this (reg, lane-half) owns
    float mx = -1e30f;
#pragma unroll
    for (int t = 0; t < 4; ++t) {
      int ng = t * 16 + nl;
      float v = s[t][r] * tauv + bs_m[mg * 64 + ng] + ms_m[mg * 64 + ng];
      s[t][r] = v;
      mx = fmaxf(mx, v);
    }
#pragma unroll
    for (int m = 1; m < 16; m <<= 1) mx = fmaxf(mx, __shfl_xor(mx, m, 32));
    float sum = 0.f;
#pragma unroll
    for (int t = 0; t < 4; ++t) {
      float e = __expf(s[t][r] - mx);
      s[t][r] = e;
      sum += e;
    }
#pragma unroll
    for (int m = 1; m < 16; m <<= 1) sum += __shfl_xor(sum, m, 32);
    float inv = 1.f / sum;
#pragma unroll
    for (int t = 0; t < 4; ++t)
      ps[(wv * 16 + r + 8 * hf) * 68 + t * 16 + nl] = (_Float16)(s[t][r] * inv);
  }
  // ps region is private to this wave; same-wave LDS RAW is in-order — no block barrier.

  // ---- O = P @ V : 2 K-chunks (tokens) x 2 head-dim tiles ----
  v8f o[2];
#pragma unroll
  for (int dt = 0; dt < 2; ++dt) o[dt] = vzero8();
#pragma unroll
  for (int c = 0; c < 2; ++c) {
    v16h ap = frag_a(ps + wv * 16 * 68 + c * 32, 68, lane);
#pragma unroll
    for (int dt = 0; dt < 2; ++dt) {
      v16h bvv = frag_b(vt + dt * 16 * 72 + c * 32, 72, lane);
      o[dt] = wmma_f16(ap, bvv, o[dt]);
    }
  }

  // ---- store (B_w, N, C) f16 for the projection GEMM ----
#pragma unroll
  for (int dt = 0; dt < 2; ++dt) {
#pragma unroll
    for (int r = 0; r < 8; ++r) {
      int mg = wv * 16 + r + 8 * hf;
      int d  = dt * 16 + nl;
      ao[((size_t)win * NTOK + mg) * CH + h * HD + d] = (_Float16)o[dt][r];
    }
  }
}

// ---------------- launcher ----------------
extern "C" void kernel_launch(void* const* d_in, const int* in_sizes, int n_in,
                              void* d_out, int out_size, void* d_ws, size_t ws_size,
                              hipStream_t stream) {
  (void)in_sizes; (void)n_in; (void)out_size; (void)ws_size;
  const float* x      = (const float*)d_in[0];
  const float* amask  = (const float*)d_in[1];
  const float* qkv_w  = (const float*)d_in[2];
  const float* qkv_b  = (const float*)d_in[3];
  const float* proj_w = (const float*)d_in[4];
  const float* proj_b = (const float*)d_in[5];
  const float* tau    = (const float*)d_in[6];
  const float* w1     = (const float*)d_in[7];
  const float* b1     = (const float*)d_in[8];
  const float* w2     = (const float*)d_in[9];
  const float* b2     = (const float*)d_in[10];

  char* ws = (char*)d_ws;
  float*    bias_ws = (float*)(ws + BIAS_OFF);
  _Float16* qkv_ws  = (_Float16*)(ws + QKV_OFF);
  _Float16* ao_ws   = (_Float16*)(ws + AO_OFF);

  // 1) relative-position-bias MLP (once, shared by all windows)
  crpb_bias_kernel<<<(NTOK * NTOK + 127) / 128, 128, 0, stream>>>(w1, b1, w2, b2, bias_ws);

  // 2) QKV projection: (262144 x 384) @ (384 x 1152) + b -> f16 workspace
  gemm64_kernel<false, true><<<dim3(C3 / 64, BW * NTOK / 64), 128, 0, stream>>>(
      x, qkv_w, qkv_b, qkv_ws, CH, C3);

  // 3) fused cosine-sim attention per (head, window)
  attn_kernel<<<dim3(NH, BW), 128, 0, stream>>>(qkv_ws, amask, bias_ws, tau, ao_ws);

  // 4) output projection: (262144 x 384) @ (384 x 384) + b -> fp32 d_out
  gemm64_kernel<true, false><<<dim3(CH / 64, BW * NTOK / 64), 128, 0, stream>>>(
      ao_ws, proj_w, proj_b, d_out, CH, CH);
}
